// TransformerModel_57853209477580
// MI455X (gfx1250) — matrix operand order repaired and checked by reading, not compiled, not run on hardware
//
#include <hip/hip_runtime.h>
#include <hip/hip_bf16.h>

// ---------------------------------------------------------------------------
// Types for CDNA5 (gfx1250) WMMA: v_wmma_f32_16x16x32_bf16
// ---------------------------------------------------------------------------
typedef __attribute__((ext_vector_type(16))) __bf16 v16bf;
typedef __attribute__((ext_vector_type(8)))  __bf16 v8bf;
typedef __attribute__((ext_vector_type(8)))  float  v8f;
typedef int v4i_vs __attribute__((vector_size(4 * sizeof(int))));

#define DIM_B   16
#define DIM_L   512
#define DIM_D   1024
#define DIM_H   16
#define DIM_DK  64
#define DIM_F   4096
#define DIM_C   10
#define DIM_NL  6

// GEMM blocking
#define BM   128
#define BN   128
#define BK   32
#define BKP  40        // padded LDS row stride (20 dwords -> conflict-free frags)

// ---------------------------------------------------------------------------
// CDNA5 async global->LDS copy path (ASYNCcnt-tracked), with a plain
// load + ds_store fallback when the builtin is unavailable.
// Probe-learned signature: (int4 __device__* src, int4 __shared__* dst, Ii, Ii)
// ---------------------------------------------------------------------------
#if defined(__has_builtin)
#  if __has_builtin(__builtin_amdgcn_global_load_async_to_lds_b128)
#    define USE_ASYNC_LDS 1
#  endif
#endif
#ifndef USE_ASYNC_LDS
#  define USE_ASYNC_LDS 0
#endif

__device__ __forceinline__ void cp16(__bf16* l, const __bf16* g) {
#if USE_ASYNC_LDS
  __builtin_amdgcn_global_load_async_to_lds_b128(
      (__attribute__((address_space(1))) v4i_vs*)(g),
      (__attribute__((address_space(3))) v4i_vs*)(l), 0, 0);
#else
  *(v8bf*)l = *(const v8bf*)g;
#endif
}

__device__ __forceinline__ void async_join() {
#if USE_ASYNC_LDS
#  if __has_builtin(__builtin_amdgcn_s_wait_asynccnt)
  __builtin_amdgcn_s_wait_asynccnt(0);
#  else
  asm volatile("s_wait_asynccnt 0x0" ::: "memory");
#  endif
#endif
}

// ---------------------------------------------------------------------------
// Fragment loaders following the CDNA5 ISA §7.12.2 register layouts.
// A (16x32 bf16): lanes 0-15 -> row M=lane, elems [0..7]=K0..7, [8..15]=K16..23
//                 lanes 16-31 -> row M=lane-16, [0..7]=K8..15, [8..15]=K24..31
// B (32x16 bf16): lanes 0-15 -> col N=lane, elems [0..7]=K0..7, [8..15]=K16..23
//                 lanes 16-31 -> col N=lane-16, [0..7]=K8..15, [8..15]=K24..31
// C/D (16x16 f32): VGPR r -> M = r (+8 for lanes>=16), N = lane&15
// ---------------------------------------------------------------------------
__device__ __forceinline__ v16bf load_a_frag(const __bf16* __restrict__ A,
                                             int lda, int row, int k0, int lane) {
  const __bf16* p = A + (size_t)row * lda + k0 + ((lane & 16) ? 8 : 0);
  union { v16bf v; v8bf h[2]; } u;
  u.h[0] = *(const v8bf*)(p);
  u.h[1] = *(const v8bf*)(p + 16);
  return u.v;
}

__device__ __forceinline__ v16bf load_b_frag(const __bf16* __restrict__ Wt,
                                             int ldb, int col, int k0, int lane) {
  const __bf16* p = Wt + (size_t)col * ldb + k0 + ((lane & 16) ? 8 : 0);
  union { v16bf v; v8bf h[2]; } u;
  u.h[0] = *(const v8bf*)(p);
  u.h[1] = *(const v8bf*)(p + 16);
  return u.v;
}

// Same layouts, reading a k-slice tile staged in LDS with padded row stride.
__device__ __forceinline__ v16bf lds_a_frag(const __bf16* rowbase, int lane) {
  const __bf16* p = rowbase + ((lane & 16) ? 8 : 0);
  union { v16bf v; v8bf h[2]; } u;
  u.h[0] = *(const v8bf*)(p);
  u.h[1] = *(const v8bf*)(p + 16);
  return u.v;
}

__device__ __forceinline__ v16bf lds_b_frag(const __bf16* rowbase, int lane) {
  const __bf16* p = rowbase + ((lane & 16) ? 8 : 0);
  union { v16bf v; v8bf h[2]; } u;
  u.h[0] = *(const v8bf*)(p);
  u.h[1] = *(const v8bf*)(p + 16);
  return u.v;
}

// ---------------------------------------------------------------------------
// LDS-tiled, double-buffered bf16 WMMA GEMM:
//   out[M,N] = A[M,K] * Wt[N,K]^T + bias[N]
// block = 256 threads = 8 waves (2 m-halves x 4 n-quarters),
// block tile 128x128, wave tile 64x32 (4x2 accumulators), k-slice 32.
// grid = (N/128, M/128).
// store modes:
//   0: f32 row-major [M,N]
//   1: bf16 scatter to Q/K layout  [H, L, DK]
//   2: bf16 scatter to V^T layout  [H, DK, L]
//   3: bf16 row-major [M,N]
// ---------------------------------------------------------------------------
__global__ __launch_bounds__(256)
void gemm_bf16_wmma(const __bf16* __restrict__ A,
                    const __bf16* __restrict__ Wt,
                    const float* __restrict__ bias,
                    float* __restrict__ outf,
                    __bf16* __restrict__ outb,
                    int M, int N, int K, int mode, int Lc) {
  __shared__ __bf16 As[2][BM * BKP];
  __shared__ __bf16 Bs[2][BN * BKP];

  const int tid  = threadIdx.x;
  const int lane = tid & 31;
  const int wave = tid >> 5;
  const int l15  = lane & 15;
  const int wm   = wave & 1;           // 64-row half
  const int wn   = wave >> 1;          // 32-col quarter
  const int m0   = blockIdx.y * BM;
  const int n0   = blockIdx.x * BN;

  // staging: thread t copies 32B of A and 32B of B per k-slice
  const int srow = tid >> 1;           // 0..127
  const int soff = (tid & 1) * 16;     // element offset in 32-elem row
  const __bf16* gArow = A  + (size_t)(m0 + srow) * K + soff;
  const __bf16* gBrow = Wt + (size_t)(n0 + srow) * K + soff;
  __bf16* sArow0 = &As[0][srow * BKP + soff];
  __bf16* sBrow0 = &Bs[0][srow * BKP + soff];
  __bf16* sArow1 = &As[1][srow * BKP + soff];
  __bf16* sBrow1 = &Bs[1][srow * BKP + soff];

  auto stage = [&](int b, int k0) {
    __bf16* sa = b ? sArow1 : sArow0;
    __bf16* sb = b ? sBrow1 : sBrow0;
    cp16(sa,     gArow + k0);
    cp16(sa + 8, gArow + k0 + 8);
    cp16(sb,     gBrow + k0);
    cp16(sb + 8, gBrow + k0 + 8);
  };

  v8f acc[4][2] = {};

  const int KT = K >> 5;
  stage(0, 0);
  async_join();
  __syncthreads();

  int buf = 0;
  for (int kt = 0; kt < KT; ++kt) {
    if (kt + 1 < KT) stage(buf ^ 1, (kt + 1) << 5);

    const __bf16* Ab = As[buf];
    const __bf16* Bb = Bs[buf];
    v16bf af[4], bfr[2];
    #pragma unroll
    for (int i = 0; i < 4; ++i)
      af[i] = lds_a_frag(Ab + (wm * 64 + i * 16 + l15) * BKP, lane);
    #pragma unroll
    for (int j = 0; j < 2; ++j)
      bfr[j] = lds_b_frag(Bb + (wn * 32 + j * 16 + l15) * BKP, lane);
    #pragma unroll
    for (int i = 0; i < 4; ++i)
      #pragma unroll
      for (int j = 0; j < 2; ++j)
        acc[i][j] = __builtin_amdgcn_wmma_f32_16x16x32_bf16(
            false, af[i], false, bfr[j], (short)0, acc[i][j], false, false);

    async_join();
    __syncthreads();
    buf ^= 1;
  }

  // epilogue
  const int mrow0 = m0 + wm * 64 + ((lane & 16) ? 8 : 0);
  #pragma unroll
  for (int j = 0; j < 2; ++j) {
    const int n = n0 + wn * 32 + j * 16 + l15;
    const float bv = bias[n];
    const int h  = n >> 6;
    const int dk = n & 63;
    #pragma unroll
    for (int i = 0; i < 4; ++i) {
      #pragma unroll
      for (int r = 0; r < 8; ++r) {
        const int m = mrow0 + i * 16 + r;
        const float cv = acc[i][j][r] + bv;
        if (mode == 0) {
          outf[(size_t)m * N + n] = cv;
        } else if (mode == 3) {
          outb[(size_t)m * N + n] = (__bf16)cv;
        } else if (mode == 1) {
          outb[((size_t)(h * Lc + m)) * DIM_DK + dk] = (__bf16)cv;
        } else {
          outb[((size_t)(h * DIM_DK + dk)) * Lc + m] = (__bf16)cv;
        }
      }
    }
  }
}

// ---------------------------------------------------------------------------
// Fused attention for batch 0 only (reference uses attn_out[0] broadcast).
// One wave per block handles one (head h, 16-row q-tile):
//   S[16,512] = (Q_tile · K^T) * scale   (WMMA)
//   softmax rows in LDS
//   O_tile[16,64] = P · V                (WMMA, P cvt f32->bf16 on the fly)
// Dynamic LDS: 16*512*4 = 32 KB.
// ---------------------------------------------------------------------------
__global__ void attention_kernel(const __bf16* __restrict__ Q,
                                 const __bf16* __restrict__ Km,
                                 const __bf16* __restrict__ Vt,
                                 __bf16* __restrict__ O,
                                 float scale) {
  extern __shared__ float S[];                // [16][512]
  const int lane = threadIdx.x & 31;
  const int l15  = lane & 15;
  const int hi   = lane >> 4;
  const int gw   = blockIdx.x;                // 0 .. H*(L/16)-1
  const int h    = gw >> 5;                   // 32 q-tiles per head
  const int qt   = gw & 31;

  const __bf16* Qh = Q  + (size_t)h * DIM_L * DIM_DK;
  const __bf16* Kh = Km + (size_t)h * DIM_L * DIM_DK;
  const __bf16* Vh = Vt + (size_t)h * DIM_DK * DIM_L;

  // ---- scores ----
  for (int nt = 0; nt < DIM_L / 16; ++nt) {
    v8f acc = {};
    #pragma unroll
    for (int k0 = 0; k0 < DIM_DK; k0 += 32) {
      v16bf a = load_a_frag(Qh, DIM_DK, qt * 16 + l15, k0, lane);
      v16bf b = load_b_frag(Kh, DIM_DK, nt * 16 + l15, k0, lane);
      acc = __builtin_amdgcn_wmma_f32_16x16x32_bf16(false, a, false, b,
                                                    (short)0, acc, false, false);
    }
    #pragma unroll
    for (int r = 0; r < 8; ++r)
      S[(hi * 8 + r) * DIM_L + nt * 16 + l15] = acc[r] * scale;
  }
  __syncthreads();

  // ---- softmax: lane pair (l15, l15+16) shares row l15, each scans 256 ----
  float* Srow = S + l15 * DIM_L;
  const int j0 = hi * 256, j1 = j0 + 256;
  float mx = -3.0e38f;
  for (int j = j0; j < j1; ++j) mx = fmaxf(mx, Srow[j]);
  mx = fmaxf(mx, __shfl_xor(mx, 16, 32));
  float sum = 0.0f;
  for (int j = j0; j < j1; ++j) {
    float e = __expf(Srow[j] - mx);
    Srow[j] = e;
    sum += e;
  }
  sum += __shfl_xor(sum, 16, 32);
  float inv = 1.0f / sum;
  for (int j = j0; j < j1; ++j) Srow[j] *= inv;
  __syncthreads();

  // ---- O = P @ V ----
  for (int nt = 0; nt < DIM_DK / 16; ++nt) {
    v8f acc = {};
    for (int k0 = 0; k0 < DIM_L; k0 += 32) {
      union { v16bf v; __bf16 e[16]; } u;
      const float* pr = S + l15 * DIM_L + k0 + ((lane & 16) ? 8 : 0);
      #pragma unroll
      for (int j = 0; j < 8; ++j) u.e[j]     = (__bf16)pr[j];
      #pragma unroll
      for (int j = 0; j < 8; ++j) u.e[8 + j] = (__bf16)pr[16 + j];
      v16bf b = load_b_frag(Vh, DIM_L, nt * 16 + l15, k0, lane);
      acc = __builtin_amdgcn_wmma_f32_16x16x32_bf16(false, u.v, false, b,
                                                    (short)0, acc, false, false);
    }
    #pragma unroll
    for (int r = 0; r < 8; ++r) {
      int l = qt * 16 + (hi ? 8 : 0) + r;
      int n = h * DIM_DK + nt * 16 + l15;
      O[(size_t)l * DIM_D + n] = (__bf16)acc[r];
    }
  }
}

// ---------------------------------------------------------------------------
// Helpers: block reduction, LayerNorm, embed, transpose, pool, classifier
// ---------------------------------------------------------------------------
__device__ __forceinline__ float block_sum256(float v) {
  __shared__ float sbuf[8];
  #pragma unroll
  for (int o = 16; o > 0; o >>= 1) v += __shfl_xor(v, o, 32);
  const int lane = threadIdx.x & 31, wave = threadIdx.x >> 5;
  if (lane == 0) sbuf[wave] = v;
  __syncthreads();
  if (threadIdx.x < 8) {
    float t = sbuf[threadIdx.x];
    #pragma unroll
    for (int o = 4; o > 0; o >>= 1) t += __shfl_xor(t, o, 32);
    if (threadIdx.x == 0) sbuf[0] = t;
  }
  __syncthreads();
  float r = sbuf[0];
  __syncthreads();
  return r;
}

// x_row = LN(x_row + add_row) * g + b  -> writes f32 x and bf16 xb (in place ok)
__global__ void add_ln_kernel(float* __restrict__ x,
                              const float* __restrict__ add, int broadcastL,
                              const float* __restrict__ g,
                              const float* __restrict__ be,
                              __bf16* __restrict__ xb) {
  const int m = blockIdx.x;                       // 0 .. B*L-1
  float* xr = x + (size_t)m * DIM_D;
  const float* ar = add + (size_t)(broadcastL ? (m & (DIM_L - 1)) : m) * DIM_D;

  float s = 0.0f;
  for (int d = threadIdx.x; d < DIM_D; d += 256) s += xr[d] + ar[d];
  const float mean = block_sum256(s) * (1.0f / DIM_D);

  float v = 0.0f;
  for (int d = threadIdx.x; d < DIM_D; d += 256) {
    float t = xr[d] + ar[d] - mean;
    v += t * t;
  }
  const float rstd = rsqrtf(block_sum256(v) * (1.0f / DIM_D) + 1e-5f);

  for (int d = threadIdx.x; d < DIM_D; d += 256) {
    float y = (xr[d] + ar[d] - mean) * rstd * g[d] + be[d];
    xr[d] = y;
    xb[(size_t)m * DIM_D + d] = (__bf16)y;
  }
}

__global__ void embed_kernel(const int* __restrict__ ids,
                             const float* __restrict__ emb,
                             const float* __restrict__ pe,
                             float* __restrict__ x, __bf16* __restrict__ xb) {
  const int m = blockIdx.x;                        // b*L + l
  const int l = m & (DIM_L - 1);
  const int id = ids[m];
  const float* er = emb + (size_t)id * DIM_D;
  const float* pr = pe + (size_t)l * DIM_D;
  const float sq = 32.0f;                          // sqrt(1024)
  for (int d = threadIdx.x; d < DIM_D; d += 256) {
    float v = er[d] * sq + pr[d];
    x[(size_t)m * DIM_D + d] = v;
    xb[(size_t)m * DIM_D + d] = (__bf16)v;
  }
}

// W [K,N] f32 row-major  ->  Wt [N,K] bf16
__global__ void transpose_to_bf16(const float* __restrict__ W,
                                  __bf16* __restrict__ Wt, int Kdim, int Ndim) {
  const int idx = blockIdx.x * 256 + threadIdx.x;
  if (idx >= Kdim * Ndim) return;
  const int k = idx / Ndim, n = idx % Ndim;
  Wt[(size_t)n * Kdim + k] = (__bf16)W[idx];
}

__global__ void pool_kernel(const float* __restrict__ x, float* __restrict__ pooled) {
  const int b = blockIdx.x;
  for (int d = threadIdx.x; d < DIM_D; d += 256) {
    float s = 0.0f;
    for (int l = 0; l < DIM_L; ++l)
      s += x[((size_t)b * DIM_L + l) * DIM_D + d];
    pooled[b * DIM_D + d] = s * (1.0f / DIM_L);
  }
}

__global__ void cls_kernel(const float* __restrict__ pooled,
                           const float* __restrict__ wf,
                           const float* __restrict__ bf_,
                           float* __restrict__ out) {
  const int idx = threadIdx.x;
  if (idx < DIM_B * DIM_C) {
    const int b = idx / DIM_C, c = idx % DIM_C;
    float s = bf_[c];
    for (int d = 0; d < DIM_D; ++d)
      s += pooled[b * DIM_D + d] * wf[d * DIM_C + c];
    out[idx] = s;
  }
}

// ---------------------------------------------------------------------------
// Host-side orchestration
// ---------------------------------------------------------------------------
extern "C" void kernel_launch(void* const* d_in, const int* in_sizes, int n_in,
                              void* d_out, int out_size, void* d_ws, size_t ws_size,
                              hipStream_t stream) {
  const int*   ids = (const int*)  d_in[0];
  const float* emb = (const float*)d_in[1];
  const float* pe  = (const float*)d_in[2];
  const float* wq  = (const float*)d_in[3];  const float* bq = (const float*)d_in[4];
  const float* wk  = (const float*)d_in[5];  const float* bk = (const float*)d_in[6];
  const float* wv  = (const float*)d_in[7];  const float* bv = (const float*)d_in[8];
  const float* wo  = (const float*)d_in[9];  const float* bo = (const float*)d_in[10];
  const float* w1  = (const float*)d_in[11]; const float* b1 = (const float*)d_in[12];
  const float* w2  = (const float*)d_in[13]; const float* b2 = (const float*)d_in[14];
  const float* g1  = (const float*)d_in[15]; const float* be1 = (const float*)d_in[16];
  const float* g2  = (const float*)d_in[17]; const float* be2 = (const float*)d_in[18];
  const float* wf  = (const float*)d_in[19]; const float* bf_ = (const float*)d_in[20];
  float* out = (float*)d_out;

  // workspace carve-up
  char* p = (char*)d_ws;
  auto takef = [&](size_t n) { float* r = (float*)p;  p += n * 4; return r; };
  auto takeb = [&](size_t n) { __bf16* r = (__bf16*)p; p += n * 2; return r; };

  const size_t ML = (size_t)DIM_B * DIM_L;           // 8192 rows
  float*  x    = takef(ML * DIM_D);                  // 32 MB
  __bf16* xb   = takeb(ML * DIM_D);                  // 16 MB
  __bf16* hb   = takeb(ML * DIM_F);                  // 64 MB
  float*  ff   = takef(ML * DIM_D);                  // 32 MB
  float*  src2 = takef((size_t)DIM_L * DIM_D);       //  2 MB
  __bf16* qb   = takeb((size_t)DIM_H * DIM_L * DIM_DK);
  __bf16* kb   = takeb((size_t)DIM_H * DIM_L * DIM_DK);
  __bf16* vtb  = takeb((size_t)DIM_H * DIM_DK * DIM_L);
  __bf16* ob   = takeb((size_t)DIM_L * DIM_D);
  __bf16* wqT  = takeb((size_t)DIM_D * DIM_D);
  __bf16* wkT  = takeb((size_t)DIM_D * DIM_D);
  __bf16* wvT  = takeb((size_t)DIM_D * DIM_D);
  __bf16* woT  = takeb((size_t)DIM_D * DIM_D);
  __bf16* w1T  = takeb((size_t)DIM_D * DIM_F);
  __bf16* w2T  = takeb((size_t)DIM_F * DIM_D);
  float*  pooled = takef((size_t)DIM_B * DIM_D);
  (void)ws_size; (void)in_sizes; (void)n_in; (void)out_size;

  // weight conversion/transposition (bf16, [N,K])
  {
    int n1 = DIM_D * DIM_D / 256;
    transpose_to_bf16<<<n1, 256, 0, stream>>>(wq, wqT, DIM_D, DIM_D);
    transpose_to_bf16<<<n1, 256, 0, stream>>>(wk, wkT, DIM_D, DIM_D);
    transpose_to_bf16<<<n1, 256, 0, stream>>>(wv, wvT, DIM_D, DIM_D);
    transpose_to_bf16<<<n1, 256, 0, stream>>>(wo, woT, DIM_D, DIM_D);
    int n2 = DIM_D * DIM_F / 256;
    transpose_to_bf16<<<n2, 256, 0, stream>>>(w1, w1T, DIM_D, DIM_F);
    transpose_to_bf16<<<n2, 256, 0, stream>>>(w2, w2T, DIM_F, DIM_D);
  }

  // x = emb[ids]*sqrt(D) + pe
  embed_kernel<<<(int)ML, 256, 0, stream>>>(ids, emb, pe, x, xb);

  const float scale = 0.125f;                        // 1/sqrt(64)
  for (int layer = 0; layer < DIM_NL; ++layer) {
    // Q,K,V projections for batch 0 only (rows 0..511 of xb)
    dim3 gqkv(DIM_D / BN, DIM_L / BM);               // (8, 4)
    gemm_bf16_wmma<<<gqkv, 256, 0, stream>>>(xb, wqT, bq, nullptr, qb,
                                             DIM_L, DIM_D, DIM_D, 1, DIM_L);
    gemm_bf16_wmma<<<gqkv, 256, 0, stream>>>(xb, wkT, bk, nullptr, kb,
                                             DIM_L, DIM_D, DIM_D, 1, DIM_L);
    gemm_bf16_wmma<<<gqkv, 256, 0, stream>>>(xb, wvT, bv, nullptr, vtb,
                                             DIM_L, DIM_D, DIM_D, 2, DIM_L);

    // fused attention (batch 0), one wave per (head, q-tile)
    attention_kernel<<<DIM_H * (DIM_L / 16), 32, 16 * DIM_L * 4, stream>>>(
        qb, kb, vtb, ob, scale);

    // src2 = O @ wo + bo  (f32 [L,D])
    gemm_bf16_wmma<<<gqkv, 256, 0, stream>>>(ob, woT, bo, src2, nullptr,
                                             DIM_L, DIM_D, DIM_D, 0, DIM_L);

    // x = LN(x + src2 broadcast)
    add_ln_kernel<<<(int)ML, 256, 0, stream>>>(x, src2, 1, g1, be1, xb);

    // FFN: hb = xb @ w1 + b1 (bf16), ff = hb @ w2 + b2 (f32)
    dim3 gf1(DIM_F / BN, (int)(ML / BM));            // (32, 64)
    gemm_bf16_wmma<<<gf1, 256, 0, stream>>>(xb, w1T, b1, nullptr, hb,
                                            (int)ML, DIM_F, DIM_D, 3, DIM_L);
    dim3 gf2(DIM_D / BN, (int)(ML / BM));            // (8, 64)
    gemm_bf16_wmma<<<gf2, 256, 0, stream>>>(hb, w2T, b2, ff, nullptr,
                                            (int)ML, DIM_D, DIM_F, 0, DIM_L);

    // x = LN(x + ff)
    add_ln_kernel<<<(int)ML, 256, 0, stream>>>(x, ff, 0, g2, be2, xb);
  }

  // pooled mean over L, then classifier
  pool_kernel<<<DIM_B, 256, 0, stream>>>(x, pooled);
  cls_kernel<<<1, 256, 0, stream>>>(pooled, wf, bf_, out);
}
